// GCNEdgeClassifier_13829794693458
// MI455X (gfx1250) — matrix operand order repaired and checked by reading, not compiled, not run on hardware
//
#include <hip/hip_runtime.h>

#define N_NODES 50000
#define N_EDGES 1600000
#define CH 128
#define N_LAYERS 8

typedef __attribute__((ext_vector_type(16))) __bf16 v16bf;
typedef __attribute__((ext_vector_type(8)))  float  v8f;

union V16U { v16bf v; unsigned int u[8]; };

__device__ __forceinline__ unsigned short f2bf(float f) {
  unsigned int u = __float_as_uint(f);
  u += 0x7FFFu + ((u >> 16) & 1u);   // round-to-nearest-even
  return (unsigned short)(u >> 16);
}
__device__ __forceinline__ unsigned int f2bf_pk(float lo, float hi) {
  return (unsigned int)f2bf(lo) | ((unsigned int)f2bf(hi) << 16);
}
__device__ __forceinline__ float bf2f(unsigned short s) {
  return __uint_as_float(((unsigned int)s) << 16);
}
__device__ __forceinline__ v8f wmma_bf16(v16bf a, v16bf b, v8f c) {
  return __builtin_amdgcn_wmma_f32_16x16x32_bf16(false, a, false, b, (short)0, c,
                                                 false, false);
}

// ---------------------------------------------------------------------------
// Weight prep: f32 -> bf16, transposed to [n][k] so B-operand K-runs are
// contiguous dwords per lane.
// ---------------------------------------------------------------------------
__global__ void prep_weights(const float* __restrict__ convW,
                             const float* __restrict__ fc0W,
                             const float* __restrict__ fc1W,
                             unsigned short* __restrict__ wtConv,
                             unsigned short* __restrict__ wtFc0,
                             unsigned short* __restrict__ wtFc1) {
  int i = blockIdx.x * 256 + threadIdx.x;
  const int NC = N_LAYERS * CH * CH;      // 131072
  if (i < NC) {
    int l = i >> 14, rem = i & 16383, k = rem >> 7, n = rem & 127;
    wtConv[l * CH * CH + n * CH + k] = f2bf(convW[i]);
  } else if (i < NC + 2 * CH * CH) {
    int j = i - NC;                       // fc0_W flat [256][128]
    int k = j >> 7, n = j & 127;
    wtFc0[n * (2 * CH) + k] = f2bf(fc0W[j]);
  } else if (i < NC + 3 * CH * CH) {
    int j = i - NC - 2 * CH * CH;         // fc1_W flat [128][128]
    int k = j >> 7, n = j & 127;
    wtFc1[n * CH + k] = f2bf(fc1W[j]);
  }
}

// ---------------------------------------------------------------------------
// Degree / normalization
// ---------------------------------------------------------------------------
__global__ void deg_init(float* __restrict__ deg) {
  int i = blockIdx.x * 256 + threadIdx.x;
  if (i < N_NODES) deg[i] = 1.0f;                 // self-loop
}
__global__ void deg_accum(const int* __restrict__ col, float* __restrict__ deg) {
  int e = blockIdx.x * 256 + threadIdx.x;
  if (e < N_EDGES) atomicAdd(&deg[col[e]], 1.0f);
}
__global__ void deg_rsqrt(float* __restrict__ deg) {
  int i = blockIdx.x * 256 + threadIdx.x;
  if (i < N_NODES) deg[i] = rsqrtf(deg[i]);
}

// vectorized: 4 channels per thread
__global__ void cvt_in(const float* __restrict__ x, unsigned int* __restrict__ hb) {
  int i = blockIdx.x * 256 + threadIdx.x;        // i indexes groups of 4
  if (i < N_NODES * CH / 4) {
    const float4 v = *(const float4*)(x + (size_t)i * 4);
    uint2 p;
    p.x = f2bf_pk(v.x, v.y);
    p.y = f2bf_pk(v.z, v.w);
    *(uint2*)(hb + (size_t)i * 2 / 2 * 2) = p;   // hb is uint*, 2 dwords per group
  }
}

// ---------------------------------------------------------------------------
// Node GEMM: hW[M][128] = bf16(h)[M][128] @ W[128][128]   (W given as Wt[n][k])
// Block: 256 threads = 8 waves, 128 rows per block; wave w owns row-tile w and
// sweeps the 8 column tiles; K = 128 -> 4 WMMAs per tile.
// ---------------------------------------------------------------------------
__global__ __launch_bounds__(256)
void gemm_node(const unsigned short* __restrict__ Abf,
               const unsigned short* __restrict__ Wt,
               float* __restrict__ out, int M) {
  int wave = threadIdx.x >> 5;
  int lane = threadIdx.x & 31;
  int rowBase = blockIdx.x * 128 + wave * 16;
  int m = rowBase + (lane & 15);
  int mA = (m < M) ? m : (M - 1);
  const unsigned int* Arow = (const unsigned int*)(Abf + (size_t)mA * CH); // 64 dw

  V16U a[4];
  int aoff = (lane < 16) ? 0 : 8;
#pragma unroll
  for (int kk = 0; kk < 4; ++kk)
#pragma unroll
    for (int q = 0; q < 8; ++q)
      a[kk].u[q] = Arow[kk * 16 + q + aoff];

  int boff = (lane < 16) ? 0 : 8;
  bool fullTile = (rowBase + 16) <= M;           // wave-uniform
#pragma unroll
  for (int nt = 0; nt < 8; ++nt) {
    int colc = nt * 16 + (lane & 15);
    const unsigned int* Bp = (const unsigned int*)(Wt + (size_t)colc * CH); // 64 dw
    v8f acc = {};
#pragma unroll
    for (int kk = 0; kk < 4; ++kk) {
      V16U b;
#pragma unroll
      for (int q = 0; q < 8; ++q) b.u[q] = Bp[kk * 16 + boff + q];
      acc = wmma_bf16(a[kk].v, b.v, acc);
    }
    int rstore = rowBase + ((lane < 16) ? 0 : 8);
    if (fullTile) {
#pragma unroll
      for (int i = 0; i < 8; ++i)
        out[(size_t)(rstore + i) * CH + colc] = acc[i];
    } else {
#pragma unroll
      for (int i = 0; i < 8; ++i) {
        int r = rstore + i;
        if (r < M) out[(size_t)r * CH + colc] = acc[i];
      }
    }
  }
}

// ---------------------------------------------------------------------------
// agg = bias + hW * dis^2 (self-loop term)  [+ residual h]   (4 ch / thread)
// ---------------------------------------------------------------------------
__global__ void agg_init(const float* __restrict__ hW, const float* __restrict__ dis,
                         const float* __restrict__ bias, const float* __restrict__ hres,
                         float* __restrict__ agg, int use_resid) {
  int i = blockIdx.x * 256 + threadIdx.x;        // group of 4
  if (i < N_NODES * CH / 4) {
    int i4 = i * 4;
    int n = i4 >> 7, c = i4 & 127;
    float d = dis[n];
    float d2 = d * d;
    const float4 hw = *(const float4*)(hW + (size_t)i4);
    const float4 bv = *(const float4*)(bias + c);
    float4 v;
    v.x = bv.x + hw.x * d2;
    v.y = bv.y + hw.y * d2;
    v.z = bv.z + hw.z * d2;
    v.w = bv.w + hw.w * d2;
    if (use_resid) {
      const float4 r = *(const float4*)(hres + (size_t)i4);
      v.x += r.x; v.y += r.y; v.z += r.z; v.w += r.w;
    }
    *(float4*)(agg + (size_t)i4) = v;
  }
}

// 32 lanes per edge, 4 channels each -> float4 load + 4 f32 atomics
__global__ void edge_scatter(const float* __restrict__ hW, const float* __restrict__ dis,
                             const int* __restrict__ row, const int* __restrict__ col,
                             float* __restrict__ agg) {
  long gid = (long)blockIdx.x * 256 + threadIdx.x;
  if (gid >= (long)N_EDGES * 32) return;
  int e = (int)(gid >> 5);
  int c = (int)(gid & 31) * 4;
  int s = row[e], t = col[e];
  float coef = dis[s] * dis[t];
  const float4 v = *(const float4*)(hW + (size_t)s * CH + c);
  float* dst = agg + (size_t)t * CH + c;
  atomicAdd(dst + 0, v.x * coef);
  atomicAdd(dst + 1, v.y * coef);
  atomicAdd(dst + 2, v.z * coef);
  atomicAdd(dst + 3, v.w * coef);
}

// h = relu(agg); keep both f32 (residual) and bf16 (next GEMM A) (4 ch/thread)
__global__ void postproc(const float* __restrict__ agg, float* __restrict__ hcur,
                         unsigned int* __restrict__ hb) {
  int i = blockIdx.x * 256 + threadIdx.x;        // group of 4
  if (i < N_NODES * CH / 4) {
    float4 v = *(const float4*)(agg + (size_t)i * 4);
    v.x = v.x > 0.0f ? v.x : 0.0f;
    v.y = v.y > 0.0f ? v.y : 0.0f;
    v.z = v.z > 0.0f ? v.z : 0.0f;
    v.w = v.w > 0.0f ? v.w : 0.0f;
    *(float4*)(hcur + (size_t)i * 4) = v;
    uint2 p;
    p.x = f2bf_pk(v.x, v.y);
    p.y = f2bf_pk(v.z, v.w);
    *(uint2*)(hb + (size_t)i * 2) = p;
  }
}

// ---------------------------------------------------------------------------
// Fused edge MLP: gather cat(h[row],h[col]) -> fc0(256->128)+relu ->
// fc1(128->128)+relu -> fc2(128->1). 128 edges per block, 8 waves, 96KB LDS
// (h2 overlays the gathered features once fc0 is done).
// ---------------------------------------------------------------------------
__global__ __launch_bounds__(256)
void edge_mlp(const unsigned short* __restrict__ hbf,
              const int* __restrict__ row, const int* __restrict__ col,
              const unsigned short* __restrict__ Wt0,
              const unsigned short* __restrict__ Wt1,
              const float* __restrict__ b0, const float* __restrict__ b1,
              const float* __restrict__ W2, const float* __restrict__ b2,
              float* __restrict__ out) {
  __shared__ char smem[96 * 1024];
  unsigned int*   efeat = (unsigned int*)smem;            // [128][128] dwords (256 bf16/row)
  unsigned short* h1    = (unsigned short*)(smem + 64 * 1024); // [128][128]
  unsigned short* h2    = (unsigned short*)smem;          // overlays efeat after fc0

  int tid = threadIdx.x;
  long eBase = (long)blockIdx.x * 128;

  // ---- gather cat(h[src], h[dst]) into LDS as bf16 dwords ----
  for (int idx = tid; idx < 128 * 128; idx += 256) {
    int e = idx >> 7, p = idx & 127;
    long ge = eBase + e;
    unsigned int val = 0;
    if (ge < N_EDGES) {
      int node = (p < 64) ? row[ge] : col[ge];
      const unsigned int* src = (const unsigned int*)(hbf + (size_t)node * CH);
      val = src[p & 63];
    }
    efeat[idx] = val;
  }
  __syncthreads();

  int wave = tid >> 5, lane = tid & 31;
  int aoff = (lane < 16) ? 0 : 8;
  int boff = (lane < 16) ? 0 : 8;
  int rbase = wave * 16 + ((lane < 16) ? 0 : 8);

  // ---- fc0: M=16 edges/wave, K=256, N=128 ----
  {
    const unsigned int* Arow = efeat + (size_t)(wave * 16 + (lane & 15)) * 128;
    V16U a[8];
#pragma unroll
    for (int kk = 0; kk < 8; ++kk)
#pragma unroll
      for (int q = 0; q < 8; ++q)
        a[kk].u[q] = Arow[kk * 16 + q + aoff];
#pragma unroll
    for (int nt = 0; nt < 8; ++nt) {
      int colc = nt * 16 + (lane & 15);
      const unsigned int* Bp = (const unsigned int*)(Wt0 + (size_t)colc * (2 * CH));
      v8f acc = {};
#pragma unroll
      for (int kk = 0; kk < 8; ++kk) {
        V16U b;
#pragma unroll
        for (int q = 0; q < 8; ++q) b.u[q] = Bp[kk * 16 + boff + q];
        acc = wmma_bf16(a[kk].v, b.v, acc);
      }
      float bias = b0[colc];
#pragma unroll
      for (int i = 0; i < 8; ++i) {
        float v = acc[i] + bias;
        v = v > 0.0f ? v : 0.0f;
        h1[(size_t)(rbase + i) * CH + colc] = f2bf(v);
      }
    }
  }
  __syncthreads();

  // ---- fc1: K=128, N=128; reads h1, writes h2 (overlay region) ----
  {
    const unsigned int* Arow = (const unsigned int*)(h1 + (size_t)(wave * 16 + (lane & 15)) * CH);
    V16U a[4];
#pragma unroll
    for (int kk = 0; kk < 4; ++kk)
#pragma unroll
      for (int q = 0; q < 8; ++q)
        a[kk].u[q] = Arow[kk * 16 + q + aoff];
#pragma unroll
    for (int nt = 0; nt < 8; ++nt) {
      int colc = nt * 16 + (lane & 15);
      const unsigned int* Bp = (const unsigned int*)(Wt1 + (size_t)colc * CH);
      v8f acc = {};
#pragma unroll
      for (int kk = 0; kk < 4; ++kk) {
        V16U b;
#pragma unroll
        for (int q = 0; q < 8; ++q) b.u[q] = Bp[kk * 16 + boff + q];
        acc = wmma_bf16(a[kk].v, b.v, acc);
      }
      float bias = b1[colc];
#pragma unroll
      for (int i = 0; i < 8; ++i) {
        float v = acc[i] + bias;
        v = v > 0.0f ? v : 0.0f;
        h2[(size_t)(rbase + i) * CH + colc] = f2bf(v);
      }
    }
  }
  __syncthreads();

  // ---- fc2: 128 -> 1 dot product, one thread per edge ----
  if (tid < 128) {
    long ge = eBase + tid;
    if (ge < N_EDGES) {
      float s = b2[0];
      const unsigned short* hr = h2 + (size_t)tid * CH;
#pragma unroll 8
      for (int k = 0; k < CH; ++k) s += bf2f(hr[k]) * W2[k];
      out[ge] = s;
    }
  }
}

// ---------------------------------------------------------------------------
extern "C" void kernel_launch(void* const* d_in, const int* in_sizes, int n_in,
                              void* d_out, int out_size, void* d_ws, size_t ws_size,
                              hipStream_t stream) {
  const float* x      = (const float*)d_in[0];
  const int*   ei     = (const int*)d_in[1];
  const float* convsW = (const float*)d_in[2];
  const float* convsB = (const float*)d_in[3];
  const float* fc0W   = (const float*)d_in[4];
  const float* fc0B   = (const float*)d_in[5];
  const float* fc1W   = (const float*)d_in[6];
  const float* fc1B   = (const float*)d_in[7];
  const float* fc2W   = (const float*)d_in[8];
  const float* fc2B   = (const float*)d_in[9];
  float* out = (float*)d_out;

  const int* row = ei;
  const int* col = ei + N_EDGES;

  // workspace carve-up (all offsets multiple of 256)
  char* ws = (char*)d_ws;
  float*          dis    = (float*)(ws + 0);                    // 200000 B (pad 200192)
  float*          hcur   = (float*)(ws + 200192);               // 25.6 MB
  float*          agg    = (float*)(ws + 200192 + 25600000);    // 25.6 MB
  float*          hW     = (float*)(ws + 200192 + 51200000);    // 25.6 MB
  unsigned short* hbf    = (unsigned short*)(ws + 200192 + 76800000); // 12.8 MB
  unsigned short* wtConv = (unsigned short*)(ws + 200192 + 89600000); // 256 KB
  unsigned short* wtFc0  = (unsigned short*)(ws + 200192 + 89862144); // 64 KB
  unsigned short* wtFc1  = (unsigned short*)(ws + 200192 + 89927680); // 32 KB

  const int T = 256;
  int gPrep  = (N_LAYERS * CH * CH + 3 * CH * CH + T - 1) / T;   // 704
  int gNode  = (N_NODES + T - 1) / T;
  int gEdge  = (N_EDGES + T - 1) / T;
  int gNC4   = (N_NODES * CH / 4 + T - 1) / T;                   // vectorized grids
  int gGemm  = (N_NODES + 127) / 128;                            // 391
  long scat  = (long)N_EDGES * 32;
  int gScat  = (int)((scat + T - 1) / T);                        // 200000
  int gMlp   = (N_EDGES + 127) / 128;                            // 12500

  prep_weights<<<gPrep, T, 0, stream>>>(convsW, fc0W, fc1W, wtConv, wtFc0, wtFc1);

  deg_init <<<gNode, T, 0, stream>>>(dis);
  deg_accum<<<gEdge, T, 0, stream>>>(col, dis);
  deg_rsqrt<<<gNode, T, 0, stream>>>(dis);

  cvt_in<<<gNC4, T, 0, stream>>>(x, (unsigned int*)hbf);

  for (int l = 0; l < N_LAYERS; ++l) {
    gemm_node<<<gGemm, T, 0, stream>>>(hbf, wtConv + (size_t)l * CH * CH, hW, N_NODES);
    agg_init <<<gNC4, T, 0, stream>>>(hW, dis, convsB + (size_t)l * CH, hcur, agg, l > 0);
    edge_scatter<<<gScat, T, 0, stream>>>(hW, dis, row, col, agg);
    postproc <<<gNC4, T, 0, stream>>>(agg, hcur, (unsigned int*)hbf);
  }

  edge_mlp<<<gMlp, T, 0, stream>>>(hbf, row, col, wtFc0, wtFc1, fc0B, fc1B, fc2W, fc2B, out);
}
